// Transformer_layer_85091892068779
// MI455X (gfx1250) — matrix operand order repaired and checked by reading, not compiled, not run on hardware
//
#include <hip/hip_runtime.h>
#include <stdint.h>

#define DMODEL 128
#define NHEAD  8
#define HDIM   16
#define MBLK   64
#define LDA    136   // padded bf16 LDS row stride (272B -> rows land on distinct banks)

typedef __attribute__((ext_vector_type(16))) __bf16        v16bf;
typedef __attribute__((ext_vector_type(8)))  float         v8f;
typedef __attribute__((ext_vector_type(4)))  unsigned int  uv4;
typedef __attribute__((ext_vector_type(2)))  unsigned int  uv2;

union FragAB {
  v16bf bf;
  uv4   u4[2];
};

__device__ __forceinline__ unsigned short f32_to_bf16(float f) {
  unsigned int u = __float_as_uint(f);
  u += 0x7FFFu + ((u >> 16) & 1u);   // round-to-nearest-even
  return (unsigned short)(u >> 16);
}
__device__ __forceinline__ unsigned int pack_bf16x2(float lo, float hi) {
  return (unsigned int)f32_to_bf16(lo) | ((unsigned int)f32_to_bf16(hi) << 16);
}
// Unpack 8 bf16 (one 16B vector) to 8 floats: hi half is just a mask.
__device__ __forceinline__ void unpack_bf16x8(uv4 u, float* f) {
  #pragma unroll
  for (int i = 0; i < 4; ++i) {
    f[2 * i]     = __uint_as_float((u[i] & 0x0000FFFFu) << 16);
    f[2 * i + 1] = __uint_as_float(u[i] & 0xFFFF0000u);
  }
}
__device__ __forceinline__ void atomic_max_f32(float* addr, float val) {
  if (val >= 0.f) atomicMax((int*)addr, __float_as_int(val));
  else            atomicMin((unsigned int*)addr, __float_as_uint(val));
}
__device__ __forceinline__ void atomic_add_f32(float* addr, float val) {
  __hip_atomic_fetch_add(addr, val, __ATOMIC_RELAXED, __HIP_MEMORY_SCOPE_AGENT);
}

// ---------------------------------------------------------------------------
// Convert 7 [128x128] fp32 weight matrices to bf16 (row-major [out][in]).
// ---------------------------------------------------------------------------
__global__ void convert_weights_kernel(const float* w0, const float* w1,
                                       const float* w2, const float* w3,
                                       const float* w4, const float* w5,
                                       const float* w6, unsigned short* dst) {
  int idx = blockIdx.x * blockDim.x + threadIdx.x;
  if (idx >= 7 * DMODEL * DMODEL) return;
  const float* srcs[7] = {w0, w1, w2, w3, w4, w5, w6};
  dst[idx] = f32_to_bf16(srcs[idx >> 14][idx & 16383]);
}

__global__ void init_kernel(float* mbuf, float* denom, float* attn, int Nn) {
  int idx = blockIdx.x * blockDim.x + threadIdx.x;
  if (idx < Nn * DMODEL) attn[idx] = 0.f;
  if (idx < Nn * NHEAD) { mbuf[idx] = -__builtin_inff(); denom[idx] = 0.f; }
}

// ---------------------------------------------------------------------------
// Shared helpers for the WMMA GEMM kernels.
// Stage a 64x128 fp32 tile into LDS as bf16 (padded stride LDA).
// ---------------------------------------------------------------------------
__device__ __forceinline__ void stage_tile(unsigned short* sA, const float* X,
                                           int m0, int M, int tid) {
  #pragma unroll
  for (int i = 0; i < 8; ++i) {
    int c    = tid + i * 256;      // 2048 float4 chunks
    int row  = c >> 5;
    int col4 = c & 31;
    int gr   = m0 + row;
    float4 v = make_float4(0.f, 0.f, 0.f, 0.f);
    if (gr < M) v = *(const float4*)(X + (size_t)gr * DMODEL + col4 * 4);
    uv2 p;
    p.x = pack_bf16x2(v.x, v.y);
    p.y = pack_bf16x2(v.z, v.w);
    *(uv2*)(&sA[row * LDA + col4 * 4]) = p;
  }
}

// One 64x16 output stripe (64 rows, 16 columns) for this wave against one weight matrix.
// acc[mt] over 4 M-tiles; A from LDS, B from global bf16 weights.
__device__ __forceinline__ void wmma_gemm_128(const unsigned short* sA,
                                              const unsigned short* Wbf,
                                              int ncol, int l16, int half,
                                              v8f acc[4]) {
  #pragma unroll
  for (int k0 = 0; k0 < DMODEL; k0 += 32) {
    FragAB fbm;
    const unsigned short* wp = Wbf + (size_t)ncol * DMODEL + k0 + half * 8;
    fbm.u4[0] = *(const uv4*)(wp);
    fbm.u4[1] = *(const uv4*)(wp + 16);
    #pragma unroll
    for (int mt = 0; mt < 4; ++mt) {
      FragAB fam;
      const unsigned short* ap = &sA[(mt * 16 + l16) * LDA + k0 + half * 8];
      fam.u4[0] = *(const uv4*)(ap);
      fam.u4[1] = *(const uv4*)(ap + 16);
      acc[mt] = __builtin_amdgcn_wmma_f32_16x16x32_bf16(
          false, fam.bf, false, fbm.bf, (short)0, acc[mt], false, false);
    }
  }
}

// ---------------------------------------------------------------------------
// Generic GEMM: Y[M,128] = op( X[M,128] @ W^T + bias ).
// op: 0 = fp32 store, 1 = fp32 + SiLU, 2 = bf16 store.
// ---------------------------------------------------------------------------
__global__ __launch_bounds__(256)
void gemm128_bf16_kernel(const float* __restrict__ X,
                         const unsigned short* __restrict__ Wbf,
                         const float* __restrict__ bias,
                         void* __restrict__ Yout, int M, int op) {
  __shared__ unsigned short sA[MBLK * LDA];
  const int tid = threadIdx.x;
  const int m0  = blockIdx.x * MBLK;
  stage_tile(sA, X, m0, M, tid);
  __syncthreads();

  const int lane = tid & 31;
  const int wave = tid >> 5;
  const int half = lane >> 4;
  const int l16  = lane & 15;
  const int ncol = wave * 16 + l16;

  const float bvv = bias[ncol];
  v8f acc[4];
  #pragma unroll
  for (int mt = 0; mt < 4; ++mt)
    acc[mt] = (v8f){bvv, bvv, bvv, bvv, bvv, bvv, bvv, bvv};

  wmma_gemm_128(sA, Wbf, ncol, l16, half, acc);

  #pragma unroll
  for (int mt = 0; mt < 4; ++mt) {
    #pragma unroll
    for (int r = 0; r < 8; ++r) {
      int gm = m0 + mt * 16 + r + half * 8;     // D layout: M = r + 8*half
      if (gm < M) {
        float v = acc[mt][r];
        if (op == 1) v = v / (1.f + __expf(-v));  // SiLU
        if (op == 2) ((unsigned short*)Yout)[(size_t)gm * DMODEL + ncol] = f32_to_bf16(v);
        else         ((float*)Yout)[(size_t)gm * DMODEL + ncol] = v;
      }
    }
  }
}

// ---------------------------------------------------------------------------
// Fused node projections: Q,K,V,S = node @ {Wq,Wk,Wv,Ws}^T + b.
// Stages node tile once; runs 4 GEMMs against contiguous bf16 weights.
// ---------------------------------------------------------------------------
__global__ __launch_bounds__(256)
void gemm128_qkvs_kernel(const float* __restrict__ X,
                         const unsigned short* __restrict__ Wcat,  // [4][128][128]
                         const float* __restrict__ bq, const float* __restrict__ bk,
                         const float* __restrict__ bv, const float* __restrict__ bs,
                         float* __restrict__ Q, float* __restrict__ K,
                         float* __restrict__ V, float* __restrict__ S, int M) {
  __shared__ unsigned short sA[MBLK * LDA];
  const int tid = threadIdx.x;
  const int m0  = blockIdx.x * MBLK;
  stage_tile(sA, X, m0, M, tid);
  __syncthreads();

  const int lane = tid & 31;
  const int wave = tid >> 5;
  const int half = lane >> 4;
  const int l16  = lane & 15;
  const int ncol = wave * 16 + l16;

  const float* biases[4] = {bq, bk, bv, bs};
  float*       outs[4]   = {Q, K, V, S};

  #pragma unroll
  for (int oi = 0; oi < 4; ++oi) {
    const float bvv = biases[oi][ncol];
    v8f acc[4];
    #pragma unroll
    for (int mt = 0; mt < 4; ++mt)
      acc[mt] = (v8f){bvv, bvv, bvv, bvv, bvv, bvv, bvv, bvv};

    wmma_gemm_128(sA, Wcat + oi * DMODEL * DMODEL, ncol, l16, half, acc);

    float* Y = outs[oi];
    #pragma unroll
    for (int mt = 0; mt < 4; ++mt) {
      #pragma unroll
      for (int r = 0; r < 8; ++r) {
        int gm = m0 + mt * 16 + r + half * 8;
        if (gm < M) Y[(size_t)gm * DMODEL + ncol] = acc[mt][r];
      }
    }
  }
}

// ---------------------------------------------------------------------------
// Pass 1: alpha[e,h] = 0.25 * q[dst,h,:] . (k[src,h,:] + Eproj[e,h,:]);
//         segment max into mbuf[dst,h] via sign-trick atomics.
// Eproj is stored bf16 to halve its HBM traffic.
// ---------------------------------------------------------------------------
__global__ void edge_alpha_kernel(const int* __restrict__ src, const int* __restrict__ dst,
                                  const float* __restrict__ Qn, const float* __restrict__ Kn,
                                  const unsigned short* __restrict__ Ep16,
                                  float* __restrict__ alpha, float* __restrict__ mbuf, int E) {
  int idx = blockIdx.x * blockDim.x + threadIdx.x;
  if (idx >= E * NHEAD) return;
  int e = idx >> 3, h = idx & 7;
  int s = src[e], d = dst[e];
  const float4* q  = (const float4*)(Qn + (size_t)d * DMODEL + h * HDIM);
  const float4* k  = (const float4*)(Kn + (size_t)s * DMODEL + h * HDIM);
  const uv4*    ep = (const uv4*)(Ep16 + (size_t)e * DMODEL + h * HDIM);
  float ef[16];
  unpack_bf16x8(ep[0], ef);
  unpack_bf16x8(ep[1], ef + 8);
  float sum = 0.f;
  #pragma unroll
  for (int j = 0; j < 4; ++j) {
    float4 qv = q[j], kv = k[j];
    sum += qv.x * (kv.x + ef[4 * j + 0]) + qv.y * (kv.y + ef[4 * j + 1])
         + qv.z * (kv.z + ef[4 * j + 2]) + qv.w * (kv.w + ef[4 * j + 3]);
  }
  sum *= 0.25f;  // 1/sqrt(C=16)
  alpha[idx] = sum;
  atomic_max_f32(&mbuf[d * NHEAD + h], sum);
}

// Pass 2: ex = exp(alpha - m[dst]); denom[dst,h] += ex (in-place overwrite).
__global__ void edge_exp_kernel(const int* __restrict__ dst, float* __restrict__ alpha,
                                const float* __restrict__ mbuf, float* __restrict__ denom,
                                int E) {
  int idx = blockIdx.x * blockDim.x + threadIdx.x;
  if (idx >= E * NHEAD) return;
  int e = idx >> 3, h = idx & 7;
  int d = dst[e];
  float ex = __expf(alpha[idx] - mbuf[d * NHEAD + h]);
  alpha[idx] = ex;
  atomic_add_f32(&denom[d * NHEAD + h], ex);
}

// Pass 3: out[dst,h,:] += (ex/denom) * (V[src,h,:] + Eproj[e,h,:]).
__global__ void edge_scatter_kernel(const int* __restrict__ src, const int* __restrict__ dst,
                                    const float* __restrict__ ex, const float* __restrict__ denom,
                                    const float* __restrict__ Vn,
                                    const unsigned short* __restrict__ Ep16,
                                    float* __restrict__ out, int E) {
  int idx = blockIdx.x * blockDim.x + threadIdx.x;
  if (idx >= E * NHEAD) return;
  int e = idx >> 3, h = idx & 7;
  int s = src[e], d = dst[e];
  float a = ex[idx] / (denom[d * NHEAD + h] + 1e-16f);
  const float4* v  = (const float4*)(Vn + (size_t)s * DMODEL + h * HDIM);
  const uv4*    ep = (const uv4*)(Ep16 + (size_t)e * DMODEL + h * HDIM);
  float ef[16];
  unpack_bf16x8(ep[0], ef);
  unpack_bf16x8(ep[1], ef + 8);
  float* o = out + (size_t)d * DMODEL + h * HDIM;
  #pragma unroll
  for (int j = 0; j < 4; ++j) {
    float4 vv = v[j];
    atomic_add_f32(o + 4 * j + 0, a * (vv.x + ef[4 * j + 0]));
    atomic_add_f32(o + 4 * j + 1, a * (vv.y + ef[4 * j + 1]));
    atomic_add_f32(o + 4 * j + 2, a * (vv.z + ef[4 * j + 2]));
    atomic_add_f32(o + 4 * j + 3, a * (vv.w + ef[4 * j + 3]));
  }
}

// ---------------------------------------------------------------------------
// out = res + LayerNorm(xa (+ xb)) * g + b.  One wave (32 lanes) per node,
// 4 floats/lane, __shfl_xor tree reduction (wave32).
// ---------------------------------------------------------------------------
__global__ __launch_bounds__(256)
void add_ln_kernel(const float* __restrict__ xa, const float* __restrict__ xb,
                   const float* __restrict__ res,
                   const float* __restrict__ g, const float* __restrict__ bsh,
                   float* __restrict__ out, int Nn) {
  int node = blockIdx.x * 8 + (threadIdx.x >> 5);
  if (node >= Nn) return;
  int lane = threadIdx.x & 31;
  size_t base = (size_t)node * DMODEL + lane * 4;
  float4 x = *(const float4*)(xa + base);
  if (xb) {
    float4 y = *(const float4*)(xb + base);
    x.x += y.x; x.y += y.y; x.z += y.z; x.w += y.w;
  }
  float s  = x.x + x.y + x.z + x.w;
  float sq = x.x * x.x + x.y * x.y + x.z * x.z + x.w * x.w;
  #pragma unroll
  for (int off = 16; off > 0; off >>= 1) {
    s  += __shfl_xor(s, off);
    sq += __shfl_xor(sq, off);
  }
  float mu   = s * (1.f / 128.f);
  float var  = sq * (1.f / 128.f) - mu * mu;
  float rstd = rsqrtf(var + 1e-5f);
  float4 gv = *(const float4*)(g   + lane * 4);
  float4 bv = *(const float4*)(bsh + lane * 4);
  float4 rv = *(const float4*)(res + base);
  float4 o;
  o.x = rv.x + (x.x - mu) * rstd * gv.x + bv.x;
  o.y = rv.y + (x.y - mu) * rstd * gv.y + bv.y;
  o.z = rv.z + (x.z - mu) * rstd * gv.z + bv.z;
  o.w = rv.w + (x.w - mu) * rstd * gv.w + bv.w;
  *(float4*)(out + base) = o;
}

// ---------------------------------------------------------------------------
extern "C" void kernel_launch(void* const* d_in, const int* in_sizes, int n_in,
                              void* d_out, int out_size, void* d_ws, size_t ws_size,
                              hipStream_t stream) {
  const int*   edge_index = (const int*)d_in[0];
  const float* node_attr  = (const float*)d_in[1];
  const float* edge_attr  = (const float*)d_in[2];
  const float* Wq = (const float*)d_in[3];  const float* bq = (const float*)d_in[4];
  const float* Wk = (const float*)d_in[5];  const float* bk = (const float*)d_in[6];
  const float* Wv = (const float*)d_in[7];  const float* bv = (const float*)d_in[8];
  const float* We = (const float*)d_in[9];  const float* be = (const float*)d_in[10];
  const float* Ws = (const float*)d_in[11]; const float* bs = (const float*)d_in[12];
  const float* W1 = (const float*)d_in[13]; const float* b1 = (const float*)d_in[14];
  const float* W2 = (const float*)d_in[15]; const float* b2 = (const float*)d_in[16];
  const float* ln1_g = (const float*)d_in[17]; const float* ln1_b = (const float*)d_in[18];
  const float* ln2_g = (const float*)d_in[19]; const float* ln2_b = (const float*)d_in[20];

  const int E  = in_sizes[0] / 2;
  const int Nn = in_sizes[1] / DMODEL;
  const int* src = edge_index;
  const int* dst = edge_index + E;

  size_t off = 0;
  auto alloc = [&](size_t bytes) -> char* {
    char* p = (char*)d_ws + off;
    off += (bytes + 255) & ~(size_t)255;
    return p;
  };
  unsigned short* wbf  = (unsigned short*)alloc((size_t)7 * 16384 * 2);
  float*          Qn   = (float*)alloc((size_t)Nn * DMODEL * 4);
  float*          Kn   = (float*)alloc((size_t)Nn * DMODEL * 4);
  float*          Vn   = (float*)alloc((size_t)Nn * DMODEL * 4);
  float*          Sn   = (float*)alloc((size_t)Nn * DMODEL * 4);
  unsigned short* Ep16 = (unsigned short*)alloc((size_t)E * DMODEL * 2);
  float*          alpha = (float*)alloc((size_t)E * NHEAD * 4);
  float*          mbuf  = (float*)alloc((size_t)Nn * NHEAD * 4);
  float*          denom = (float*)alloc((size_t)Nn * NHEAD * 4);
  float*          attn  = (float*)alloc((size_t)Nn * DMODEL * 4);
  float*          node1 = (float*)alloc((size_t)Nn * DMODEL * 4);
  float*          h1    = (float*)alloc((size_t)Nn * DMODEL * 4);
  float*          h2    = (float*)alloc((size_t)Nn * DMODEL * 4);

  // Converted weight order: [wq, wk, wv, ws] (contiguous for fused kernel), we, w1, w2
  unsigned short* wqkvs = wbf;                  // 4 matrices, contiguous
  unsigned short* we    = wbf + 4 * 16384;
  unsigned short* w1    = wbf + 5 * 16384;
  unsigned short* w2    = wbf + 6 * 16384;

  convert_weights_kernel<<<(7 * 16384 + 255) / 256, 256, 0, stream>>>(
      Wq, Wk, Wv, Ws, We, W1, W2, wbf);
  init_kernel<<<(Nn * DMODEL + 255) / 256, 256, 0, stream>>>(mbuf, denom, attn, Nn);

  const int gn  = (Nn + MBLK - 1) / MBLK;
  const int ge  = (E + MBLK - 1) / MBLK;
  const int geh = (E * NHEAD + 255) / 256;
  const int gln = (Nn + 7) / 8;

  // Fused node projections (one read of node_attr, 4 WMMA GEMMs)
  gemm128_qkvs_kernel<<<gn, 256, 0, stream>>>(node_attr, wqkvs, bq, bk, bv, bs,
                                              Qn, Kn, Vn, Sn, Nn);
  // Edge projection, stored bf16 (halves Eproj HBM traffic)
  gemm128_bf16_kernel<<<ge, 256, 0, stream>>>(edge_attr, we, be, Ep16, E, 2);

  // Segment softmax + scatter
  edge_alpha_kernel<<<geh, 256, 0, stream>>>(src, dst, Qn, Kn, Ep16, alpha, mbuf, E);
  edge_exp_kernel<<<geh, 256, 0, stream>>>(dst, alpha, mbuf, denom, E);
  edge_scatter_kernel<<<geh, 256, 0, stream>>>(src, dst, alpha, denom, Vn, Ep16, attn, E);

  // Residual + LN1 (x = attn + skip)
  add_ln_kernel<<<gln, 256, 0, stream>>>(attn, Sn, node_attr, ln1_g, ln1_b, node1, Nn);

  // FFN: h1 = silu(node1 @ W1^T + b1); h2 = h1 @ W2^T + b2
  gemm128_bf16_kernel<<<gn, 256, 0, stream>>>(node1, w1, b1, h1, Nn, 1);
  gemm128_bf16_kernel<<<gn, 256, 0, stream>>>(h1, w2, b2, h2, Nn, 0);

  // Residual + LN2 -> output
  add_ln_kernel<<<gln, 256, 0, stream>>>(h2, nullptr, node1, ln2_g, ln2_b,
                                         (float*)d_out, Nn);
}
